// Mistral4MoE_67422396613250
// MI455X (gfx1250) — matrix operand order repaired and checked by reading, not compiled, not run on hardware
//
#include <hip/hip_runtime.h>
#include <hip/hip_bf16.h>

// ---------------- problem constants (match reference) ----------------
constexpr int T    = 2 * 2048;   // tokens
constexpr int H    = 4096;       // hidden
constexpr int I    = 2048;       // expert intermediate (== SI)
constexpr int E    = 16;         // routed experts
constexpr int TOPK = 4;
constexpr int TK   = T * TOPK;   // 16384 routed slots
constexpr int NSLOT = TK + T;    // + shared-expert slots (segment 16)

// GEMM tiling
constexpr int BM = 32;           // token rows per block
constexpr int BN = 64;           // output cols per block
constexpr int KC = 32;           // K chunk (WMMA bf16 K)
constexpr int LDT = 40;          // padded LDS row stride (bf16 elems), 80B, 16B-aligned

typedef __attribute__((ext_vector_type(16))) __bf16 v16bf;
typedef __attribute__((ext_vector_type(8)))  float  v8f;

union FragU { v16bf v; uint4 q[2]; };

// A matrix 16x32 bf16 (ISA 7.12.2): m=lane&15, half=lane>>4,
// VGPR e(0..3): K = half*8 + 2e,2e+1 ; VGPR e(4..7): K = 16 + half*8 + ...
__device__ inline v16bf frag_A(const __bf16* tile, int lane) {
  int m = lane & 15, half = lane >> 4;
  const __bf16* p = tile + m * LDT + half * 8;
  FragU f;
  f.q[0] = *(const uint4*)(p);
  f.q[1] = *(const uint4*)(p + 16);
  return f.v;
}

// B matrix 32x16 bf16: n=lane&15, half=lane>>4, VGPR e(0..3): K = half*8 + 2e,2e+1 ; VGPR e(4..7): K = 16 + half*8 + ...
__device__ inline v16bf frag_B(const __bf16* tile, int lane) {
  int n = lane & 15, half = lane >> 4;
  const __bf16* p = tile + n * LDT + half * 8;
  FragU f;
  f.q[0] = *(const uint4*)(p);
  f.q[1] = *(const uint4*)(p + 16);
  return f.v;
}

__device__ inline void cvt_store4(__bf16* d, float4 v) {
  d[0] = (__bf16)v.x; d[1] = (__bf16)v.y; d[2] = (__bf16)v.z; d[3] = (__bf16)v.w;
}

__device__ inline float fast_silu(float g) {
  return g * __builtin_amdgcn_rcpf(1.f + __expf(-g));
}

// async global -> LDS (gfx1250, ASYNCcnt-tracked), GV mode: 64-bit vaddr, saddr=off
__device__ inline void async_load_b64(unsigned lds_off, const void* gaddr) {
  asm volatile("global_load_async_to_lds_b64 %0, %1, off"
               :: "v"(lds_off), "v"(gaddr) : "memory");
}
__device__ inline void wait_async0() {
  asm volatile("s_wait_asynccnt 0" ::: "memory");
}

// ---------------- 1) router ----------------
__global__ __launch_bounds__(512) void router_kernel(
    const float* __restrict__ x, const float* __restrict__ gw,
    const float* __restrict__ bias, float* __restrict__ cw) {
  const int t = blockIdx.x;
  __shared__ __align__(16) float xs[H];
  __shared__ float sc[E];
  const float* xr = x + (size_t)t * H;
  for (int i = threadIdx.x; i < H / 4; i += blockDim.x)
    ((float4*)xs)[i] = ((const float4*)xr)[i];
  __syncthreads();
  const int wave = threadIdx.x >> 5, lane = threadIdx.x & 31;
  if (wave < E) {
    const float* g = gw + (size_t)wave * H;
    float s = 0.f;
    for (int j = lane; j < H; j += 32) s += xs[j] * g[j];
    for (int off = 16; off > 0; off >>= 1) s += __shfl_xor(s, off, 32);
    if (lane == 0) sc[wave] = 1.f / (1.f + __expf(-s)) + bias[wave];
  }
  __syncthreads();
  if (threadIdx.x == 0) {
    float v[E]; bool used[E];
    for (int e = 0; e < E; ++e) { v[e] = sc[e]; used[e] = false; }
    int sel[TOPK]; float wv[TOPK]; float sum = 0.f;
    for (int k = 0; k < TOPK; ++k) {
      int bi = 0; float bv = -1e30f;
      for (int e = 0; e < E; ++e)
        if (!used[e] && v[e] > bv) { bv = v[e]; bi = e; }
      used[bi] = true; sel[k] = bi; wv[k] = bv; sum += bv;
    }
    const float inv = 1.f / (sum + 1e-20f);
    float row[E];
    for (int e = 0; e < E; ++e) row[e] = 0.f;
    for (int k = 0; k < TOPK; ++k) row[sel[k]] = wv[k] * inv;
    for (int e = 0; e < E; ++e) cw[(size_t)t * E + e] = row[e];
  }
}

// ---------------- 2) compact ----------------
__global__ __launch_bounds__(512) void compact_kernel(
    const float* __restrict__ cw, int* __restrict__ counts,
    int* __restrict__ offsets, int* __restrict__ slot_tok,
    float* __restrict__ slot_w) {
  __shared__ int scnt[E];
  const int wave = threadIdx.x >> 5, lane = threadIdx.x & 31;
  if (wave < E) {
    int cnt = 0;
    for (int t0 = 0; t0 < T; t0 += 32) {
      float w = cw[(size_t)(t0 + lane) * E + wave];
      unsigned mask = __builtin_amdgcn_ballot_w32(w > 0.f);
      cnt += __builtin_popcount(mask);
    }
    if (lane == 0) scnt[wave] = cnt;
  }
  __syncthreads();
  if (threadIdx.x == 0) {
    int run = 0;
    for (int e = 0; e < E; ++e) { offsets[e] = run; counts[e] = scnt[e]; run += scnt[e]; }
    offsets[E] = TK;
    counts[E]  = T;
  }
  __syncthreads();
  if (wave < E) {
    int base = 0;
    for (int e = 0; e < wave; ++e) base += scnt[e];
    for (int t0 = 0; t0 < T; t0 += 32) {
      int t = t0 + lane;
      float w = cw[(size_t)t * E + wave];
      unsigned mask = __builtin_amdgcn_ballot_w32(w > 0.f);
      if (w > 0.f) {
        int pos = base + __builtin_popcount(mask & ((1u << lane) - 1u));
        slot_tok[pos] = t; slot_w[pos] = w;
      }
      base += __builtin_popcount(mask);
    }
  }
  for (int t = threadIdx.x; t < T; t += blockDim.x) {
    slot_tok[TK + t] = t;
    slot_w[TK + t]   = 1.f;
  }
}

// ---------------- 3) gate/up GEMM + SiLU, z -> bf16 scratch ----------------
__global__ __launch_bounds__(256) void gateup_kernel(
    const float* __restrict__ x,  const float* __restrict__ Wg,
    const float* __restrict__ Wu, const float* __restrict__ Sg,
    const float* __restrict__ Su, const int* __restrict__ counts,
    const int* __restrict__ offsets, const int* __restrict__ slot_tok,
    const float* __restrict__ slot_w, __bf16* __restrict__ zbuf) {
  constexpr int NT = I / BN;            // 32
  constexpr int MT = T / BM;            // 128
  constexpr int BPSEG = MT * NT;
  constexpr int NCH = H / KC;           // 128
  const int seg = blockIdx.x / BPSEG;
  const int rem = blockIdx.x % BPSEG;
  const int mt = rem / NT, nt = rem % NT;
  const int cnt = counts[seg];
  if (mt * BM >= cnt) return;
  const int off = offsets[seg];
  const float* WgE = (seg < E) ? Wg + (size_t)seg * I * H : Sg;
  const float* WuE = (seg < E) ? Wu + (size_t)seg * I * H : Su;

  __shared__ __align__(16) __bf16 Xs[2 * BM * LDT];
  __shared__ __align__(16) __bf16 Gs[2 * BN * LDT];
  __shared__ __align__(16) __bf16 Us[2 * BN * LDT];
  __shared__ int   ts[BM];
  __shared__ float tw[BM];

  const int tid = threadIdx.x;
  if (tid < BM) {
    int rr = mt * BM + tid;
    int p  = off + ((rr < cnt) ? rr : 0);
    ts[tid] = slot_tok[p];
    tw[tid] = slot_w[p];
  }
  __syncthreads();

  const int lane = tid & 31, wave = tid >> 5;
  const int wm = wave >> 2, wn = wave & 3;
  v8f cg = {}; v8f cu = {};

  // loader coordinates
  const int xr_ = tid >> 3;             // 0..31
  const int xc_ = (tid & 7) * 4;        // 0..28
  const int wr0 = tid >> 3;             // 0..31
  const int wr1 = 32 + wr0;             // 32..63
  const size_t xrow  = (size_t)ts[xr_] * H + xc_;
  const size_t wrow0 = (size_t)(nt * BN + wr0) * H + xc_;
  const size_t wrow1 = (size_t)(nt * BN + wr1) * H + xc_;

  // prologue: chunk 0 -> buffer 0
  {
    cvt_store4(Xs + xr_ * LDT + xc_, *(const float4*)(x + xrow));
    cvt_store4(Gs + wr0 * LDT + xc_, *(const float4*)(WgE + wrow0));
    cvt_store4(Gs + wr1 * LDT + xc_, *(const float4*)(WgE + wrow1));
    cvt_store4(Us + wr0 * LDT + xc_, *(const float4*)(WuE + wrow0));
    cvt_store4(Us + wr1 * LDT + xc_, *(const float4*)(WuE + wrow1));
  }
  __syncthreads();

  int p = 0;
  for (int it = 0; it < NCH; ++it) {
    const bool has_next = (it + 1 < NCH);
    float4 nx, ng0, ng1, nu0, nu1;
    if (has_next) {
      const int h0 = (it + 1) * KC;
      nx  = *(const float4*)(x   + xrow  + h0);
      ng0 = *(const float4*)(WgE + wrow0 + h0);
      ng1 = *(const float4*)(WgE + wrow1 + h0);
      nu0 = *(const float4*)(WuE + wrow0 + h0);
      nu1 = *(const float4*)(WuE + wrow1 + h0);
    }
    const __bf16* Xb = Xs + p * BM * LDT;
    const __bf16* Gb = Gs + p * BN * LDT;
    const __bf16* Ub = Us + p * BN * LDT;
    v16bf a  = frag_A(Xb + wm * 16 * LDT, lane);
    v16bf bg = frag_B(Gb + wn * 16 * LDT, lane);
    v16bf bu = frag_B(Ub + wn * 16 * LDT, lane);
    cg = __builtin_amdgcn_wmma_f32_16x16x32_bf16(false, a, false, bg, (short)0, cg, false, false);
    cu = __builtin_amdgcn_wmma_f32_16x16x32_bf16(false, a, false, bu, (short)0, cu, false, false);
    if (has_next) {
      const int q = p ^ 1;
      cvt_store4(Xs + q * BM * LDT + xr_ * LDT + xc_, nx);
      cvt_store4(Gs + q * BN * LDT + wr0 * LDT + xc_, ng0);
      cvt_store4(Gs + q * BN * LDT + wr1 * LDT + xc_, ng1);
      cvt_store4(Us + q * BN * LDT + wr0 * LDT + xc_, nu0);
      cvt_store4(Us + q * BN * LDT + wr1 * LDT + xc_, nu1);
    }
    __syncthreads();
    p ^= 1;
  }

  // epilogue: z = silu(g)*u*w, stage in LDS (reuse Gs), coalesced b128 store
  __bf16* zstage = Gs;   // 32x64 bf16 = 4KB
  {
    const int n = lane & 15, halfm = lane >> 4;
#pragma unroll
    for (int rg = 0; rg < 8; ++rg) {
      const int m = wm * 16 + halfm * 8 + rg;
      const float z = fast_silu(cg[rg]) * cu[rg] * tw[m];
      zstage[m * BN + wn * 16 + n] = (__bf16)z;
    }
  }
  __syncthreads();
  {
    const int srow = tid >> 3;           // 0..31
    const int scol = (tid & 7) * 8;      // 8 bf16 = 16B
    const int rr = mt * BM + srow;
    if (rr < cnt) {
      *(uint4*)(zbuf + (size_t)(off + rr) * I + nt * BN + scol) =
          *(const uint4*)(zstage + srow * BN + scol);
    }
  }
}

// ---------------- 4) down-proj GEMM, atomic scatter into out ----------------
__global__ __launch_bounds__(256) void down_kernel(
    const __bf16* __restrict__ zbuf, const float* __restrict__ Wd,
    const float* __restrict__ Sd, const int* __restrict__ counts,
    const int* __restrict__ offsets, const int* __restrict__ slot_tok,
    float* __restrict__ out) {
  constexpr int NT = H / BN;            // 64
  constexpr int MT = T / BM;            // 128
  constexpr int BPSEG = MT * NT;
  constexpr int NCH = I / KC;           // 64
  const int seg = blockIdx.x / BPSEG;
  const int rem = blockIdx.x % BPSEG;
  const int mt = rem / NT, nt = rem % NT;
  const int cnt = counts[seg];
  if (mt * BM >= cnt) return;
  const int off = offsets[seg];
  const float* WdE = (seg < E) ? Wd + (size_t)seg * H * I : Sd;

  __shared__ __align__(16) __bf16 Zs[2 * BM * LDT];
  __shared__ __align__(16) __bf16 Ds[2 * BN * LDT];
  __shared__ int ts[BM];

  const int tid = threadIdx.x;
  if (tid < BM) {
    int rr = mt * BM + tid;
    ts[tid] = slot_tok[off + ((rr < cnt) ? rr : 0)];
  }

  const int lane = tid & 31, wave = tid >> 5;
  const int wm = wave >> 2, wn = wave & 3;
  v8f c = {};

  const int zr_ = tid >> 3;             // 0..31
  const int zc_ = (tid & 7) * 4;        // 0..28 (4 bf16 = 8B)
  const int wr0 = tid >> 3;
  const int wr1 = 32 + wr0;
  const size_t zrow  = (size_t)(off + mt * BM + zr_) * I + zc_;
  const size_t wrow0 = (size_t)(nt * BN + wr0) * I + zc_;
  const size_t wrow1 = (size_t)(nt * BN + wr1) * I + zc_;

  // prologue: chunk 0 -> buffer 0 (z via async DMA to LDS, weights via cvt)
  async_load_b64((unsigned)(size_t)(Zs + zr_ * LDT + zc_), zbuf + zrow);
  cvt_store4(Ds + wr0 * LDT + zc_, *(const float4*)(WdE + wrow0));
  cvt_store4(Ds + wr1 * LDT + zc_, *(const float4*)(WdE + wrow1));
  wait_async0();
  __syncthreads();

  int p = 0;
  for (int it = 0; it < NCH; ++it) {
    const bool has_next = (it + 1 < NCH);
    float4 nd0, nd1;
    if (has_next) {
      const int i0 = (it + 1) * KC;
      const int q = p ^ 1;
      async_load_b64((unsigned)(size_t)(Zs + q * BM * LDT + zr_ * LDT + zc_),
                     zbuf + zrow + i0);
      nd0 = *(const float4*)(WdE + wrow0 + i0);
      nd1 = *(const float4*)(WdE + wrow1 + i0);
    }
    const __bf16* Zb = Zs + p * BM * LDT;
    const __bf16* Db = Ds + p * BN * LDT;
    v16bf a = frag_A(Zb + wm * 16 * LDT, lane);
    v16bf b = frag_B(Db + wn * 16 * LDT, lane);
    c = __builtin_amdgcn_wmma_f32_16x16x32_bf16(false, a, false, b, (short)0, c, false, false);
    if (has_next) {
      const int q = p ^ 1;
      cvt_store4(Ds + q * BN * LDT + wr0 * LDT + zc_, nd0);
      cvt_store4(Ds + q * BN * LDT + wr1 * LDT + zc_, nd1);
    }
    wait_async0();
    __syncthreads();
    p ^= 1;
  }

  // stage results in LDS (reuse Ds as f32), then coalesced row-wise atomics
  float* fstage = (float*)Ds;   // 32x64 f32 = 8KB <= Ds (10240B)
  {
    const int n = lane & 15, halfm = lane >> 4;
#pragma unroll
    for (int rg = 0; rg < 8; ++rg) {
      const int m = wm * 16 + halfm * 8 + rg;
      fstage[m * BN + wn * 16 + n] = c[rg];
    }
  }
  __syncthreads();
  {
    const int srow = tid >> 3;
    const int scol = (tid & 7) * 8;
    const int rr = mt * BM + srow;
    if (rr < cnt) {
      float* dst = out + (size_t)ts[srow] * H + nt * BN + scol;
      const float* src = fstage + srow * BN + scol;
#pragma unroll
      for (int j = 0; j < 8; ++j)
        __hip_atomic_fetch_add(dst + j, src[j], __ATOMIC_RELAXED, __HIP_MEMORY_SCOPE_AGENT);
    }
  }
}

// ---------------- host launcher ----------------
extern "C" void kernel_launch(void* const* d_in, const int* in_sizes, int n_in,
                              void* d_out, int out_size, void* d_ws, size_t ws_size,
                              hipStream_t stream) {
  (void)in_sizes; (void)n_in; (void)ws_size;
  const float* x    = (const float*)d_in[0];
  const float* gw   = (const float*)d_in[1];
  const float* bias = (const float*)d_in[2];
  const float* Wg   = (const float*)d_in[3];
  const float* Wu   = (const float*)d_in[4];
  const float* Wd   = (const float*)d_in[5];
  const float* Sg   = (const float*)d_in[6];
  const float* Su   = (const float*)d_in[7];
  const float* Sd   = (const float*)d_in[8];
  float* out = (float*)d_out;

  char* ws = (char*)d_ws;
  int*    counts   = (int*)(ws);
  int*    offsets  = (int*)(ws + 128);
  float*  cw       = (float*)(ws + 256);
  size_t  o        = 256 + (size_t)T * E * sizeof(float);
  int*    slot_tok = (int*)(ws + o);   o += (size_t)NSLOT * sizeof(int);
  float*  slot_w   = (float*)(ws + o); o += (size_t)NSLOT * sizeof(float);
  o = (o + 255) & ~(size_t)255;
  __bf16* zbuf     = (__bf16*)(ws + o);   // NSLOT * I bf16 (~84 MB)

  hipMemsetAsync(d_out, 0, (size_t)out_size * sizeof(float), stream);

  router_kernel<<<T, 512, 0, stream>>>(x, gw, bias, cw);
  compact_kernel<<<1, 512, 0, stream>>>(cw, counts, offsets, slot_tok, slot_w);

  const int gu_blocks = (E + 1) * (T / BM) * (I / BN);   // 17*128*32
  gateup_kernel<<<gu_blocks, 256, 0, stream>>>(x, Wg, Wu, Sg, Su, counts, offsets,
                                               slot_tok, slot_w, zbuf);

  const int dn_blocks = (E + 1) * (T / BM) * (H / BN);   // 17*128*64
  down_kernel<<<dn_blocks, 256, 0, stream>>>(zbuf, Wd, Sd, counts, offsets,
                                             slot_tok, out);
}
